// RQ_creator_conv_gru_80771154969424
// MI455X (gfx1250) — matrix operand order repaired and checked by reading, not compiled, not run on hardware
//
#include <hip/hip_runtime.h>
#include <math.h>

// ---------------------------------------------------------------------------
// MI455X (gfx1250): complex conv1d + BN -> 16 complex GRU layers -> FF ->
// mean -> dense heads + norm.  All GEMMs use V_WMMA_F32_16X16X32_F16.
// Weights are repacked once per call into WMMA *fragment-major* f16 layout so
// every B fragment is a contiguous 32B per-lane load (2x global_load_b128)
// instead of 16 strided u16 loads.  The sequential GRU recurrence runs as 4
// persistent workgroups with recurrent-weight fragments resident in VGPRs.
// ---------------------------------------------------------------------------

#define DEVI __device__ __forceinline__

typedef __attribute__((ext_vector_type(16))) _Float16 v16h;
typedef __attribute__((ext_vector_type(8)))  _Float16 v8h;
typedef __attribute__((ext_vector_type(8)))  float    v8f;
typedef _Float16 h16_t;

constexpr int B_ = 16, T_ = 384, CIN = 64, F_ = 64, KW = 32, U_ = 128;
constexpr int FF_ = 1024, NN = 4096, LS = 15;
constexpr int BT = B_ * T_;          // 6144
constexpr int TU = 3 * U_;           // 384
constexpr int NLAYER = LS + 1;       // 16
constexpr float EPS = 1e-3f;

// ---------------- workspace layout (element counts, halves first) ----------
constexpr size_t SZ_WC16 = (size_t)KW * 128 * 128;       // fused conv weights (frag)
constexpr size_t SZ_HCAT = (size_t)BT * 128;             // [hr|hi] input, f16
constexpr size_t SZ_GK_L = (size_t)2 * 128 * TU;         // per-layer (kr,ki) block
constexpr size_t SZ_GK   = (size_t)NLAYER * SZ_GK_L;
constexpr size_t SZ_GRK  = (size_t)NLAYER * SZ_GK_L;     // per-layer (rkr,rki)
constexpr size_t SZ_FF1  = (size_t)128 * FF_;
constexpr size_t SZ_FF2  = (size_t)FF_ * 128;
constexpr size_t SZ_QW1  = (size_t)128 * NN;
constexpr size_t SZ_QW   = 4 * SZ_QW1;                   // qr,qi,rr,ri
constexpr size_t SZ_X    = (size_t)BT * 128;             // xr16 / xi16
constexpr size_t SZ_XP   = 4ull * BT * TU;
constexpr size_t SZ_YS   = 4ull * BT * 128;
constexpr size_t SZ_M16  = 2ull * 16 * 128;

constexpr size_t OFF_WC16 = 0;
constexpr size_t OFF_HCAT = OFF_WC16 + SZ_WC16;
constexpr size_t OFF_GK   = OFF_HCAT + SZ_HCAT;
constexpr size_t OFF_GRK  = OFF_GK + SZ_GK;
constexpr size_t OFF_FF1  = OFF_GRK + SZ_GRK;
constexpr size_t OFF_FF2  = OFF_FF1 + SZ_FF1;
constexpr size_t OFF_QW   = OFF_FF2 + SZ_FF2;
constexpr size_t OFF_XR   = OFF_QW + SZ_QW;
constexpr size_t OFF_XI   = OFF_XR + SZ_X;
constexpr size_t OFF_XP   = OFF_XI + SZ_X;
constexpr size_t OFF_YS   = OFF_XP + SZ_XP;
constexpr size_t OFF_M16  = OFF_YS + SZ_YS;
constexpr size_t HALF_TOTAL = OFF_M16 + SZ_M16;          // ~21M halves (~42MB)

// ---------------- WMMA helpers (wave32, A/B in contiguous k order) ----------------
DEVI v8f wmma32(v16h a, v16h b, v8f c) {
  return __builtin_amdgcn_wmma_f32_16x16x32_f16(false, a, false, b,
                                                (short)0, c, false, false);
}

// A: 16x32 f16 tile from row-major src (lda halves between rows).
// lane<16: row=lane,   halves [0..15]=K0..15
// lane>=16: row=lane-16, halves [0..15]=K16..31
DEVI v16h load_a(const h16_t* src, int lda) {
  int lane = threadIdx.x & 31;
  int r = lane & 15;
  int kb = (lane >> 4) << 4;          // 0 or 16
  const h16_t* p = src + (size_t)r * lda + kb;
  v16h a;
#pragma unroll
  for (int i = 0; i < 8; ++i) { a[i] = p[i]; a[8 + i] = p[8 + i]; }
  return a;
}

// B fragment from fragment-major storage: 512 halves per 32x16 tile,
// 16 contiguous halves per lane (32B -> 2x b128 loads).
DEVI v16h load_bfrag(const h16_t* tileBase) {
  int lane = threadIdx.x & 31;
  return *(const v16h*)(tileBase + (size_t)lane * 16);
}

// C/D: 16x16 f32: N=lane&15, rows mb..mb+7 with mb = (lane>=16)*8
DEVI void store_c(float* dst, int ldc, v8f c) {
  int lane = threadIdx.x & 31;
  int n = lane & 15;
  int mb = (lane >> 4) << 3;
#pragma unroll
  for (int i = 0; i < 8; ++i) dst[(size_t)(mb + i) * ldc + n] = c[i];
}

// ---------------- weight packing to fragment-major f16 ---------------------
// Fragment element mapping for a KxN matrix (KT=K/32, NT=N/16 tiles):
//   out[((kt*NT+nt)*32 + lane)*16 + i] = w[kt*32 + (lane>=16)*16 + i][nt*16 + (lane&15)]
__global__ void pack_frag(const float* __restrict__ s, h16_t* __restrict__ d,
                          int K, int N, int nmat, size_t sStride, size_t dStride) {
  int per = K * N;
  int i = blockIdx.x * 256 + threadIdx.x;
  if (i >= per * nmat) return;
  int m = i / per, o = i % per;
  int i16 = o & 15, lane = (o >> 4) & 31, tile = o >> 9;
  int NT = N >> 4;
  int nt = tile % NT, kt = tile / NT;
  int k = kt * 32 + ((lane >> 4) << 4) + i16;
  int n = nt * 16 + (lane & 15);
  d[m * dStride + o] = (h16_t)s[m * sStride + (size_t)k * N + n];
}

// Fused complex conv weight, fragment-major per tap (128x128, NT=8, KT=4).
// cols 0..63 -> real out:  rows 0..63 = wr, rows 64..127 = -wi
// cols 64..127 -> imag out: rows 0..63 = wi, rows 64..127 =  wr
__global__ void pack_conv_frag(const float* __restrict__ wr, const float* __restrict__ wi,
                               h16_t* __restrict__ d) {
  int i = blockIdx.x * 256 + threadIdx.x;
  if (i >= KW * 128 * 128) return;
  int tap = i >> 14, o = i & 16383;
  int i16 = o & 15, lane = (o >> 4) & 31, tile = o >> 9;
  int nt = tile & 7, kt = tile >> 3;
  int r = kt * 32 + ((lane >> 4) << 4) + i16;     // input-channel row (0..127)
  int n = nt * 16 + (lane & 15);                  // output col (0..127)
  float v;
  if (n < 64) v = (r < 64) ?  wr[(tap * CIN + r) * F_ + n]
                           : -wi[(tap * CIN + (r - 64)) * F_ + n];
  else {
    int f = n - 64;
    v = (r < 64) ? wi[(tap * CIN + r) * F_ + f]
                 : wr[(tap * CIN + (r - 64)) * F_ + f];
  }
  d[i] = (h16_t)v;
}

__global__ void pack_hcat(const float* __restrict__ hr, const float* __restrict__ hi,
                          h16_t* __restrict__ d) {
  int i = blockIdx.x * 256 + threadIdx.x;
  if (i >= BT * 128) return;
  int bt = i >> 7, c = i & 127;
  d[i] = (h16_t)(c < 64 ? hr[bt * 64 + c] : hi[bt * 64 + (c - 64)]);
}

// ---------------- conv1d (SAME, per-tap GEMM) + BN --------------------------
// grid 384 (16-row M tiles), block 256 (8 waves, one 16-col N tile each)
__global__ void conv_bn_kernel(const h16_t* __restrict__ hcat, const h16_t* __restrict__ wc,
                               const float* __restrict__ gr, const float* __restrict__ br,
                               const float* __restrict__ mr, const float* __restrict__ vr,
                               const float* __restrict__ gi, const float* __restrict__ bi,
                               const float* __restrict__ mi, const float* __restrict__ vi,
                               h16_t* __restrict__ xr, h16_t* __restrict__ xi) {
  __shared__ __align__(32) h16_t atile[16 * 128];
  int mt = blockIdx.x;
  int b = mt / 24, t0 = (mt % 24) * 16;
  int tid = threadIdx.x, wv = tid >> 5, lane = tid & 31;
  int n0 = wv * 16;
  // staging coordinates: one 16B vector per thread covers 16x128 tile
  int sidx = tid * 8;
  int srow = sidx >> 7, scol = sidx & 127;
  v8f acc = {};
  for (int k = 0; k < KW; ++k) {
    int tt = t0 + srow + k - 15;                  // SAME pad (15,16)
    v8h v = {};
    if (tt >= 0 && tt < T_) v = *(const v8h*)(hcat + ((size_t)(b * T_ + tt)) * 128 + scol);
    *(v8h*)(atile + sidx) = v;
    __syncthreads();
    const h16_t* wk = wc + (size_t)k * 16384;
#pragma unroll
    for (int kt = 0; kt < 4; ++kt) {
      v16h a = load_a(atile + kt * 32, 128);
      v16h bb = load_bfrag(wk + (size_t)(kt * 8 + wv) * 512);
      acc = wmma32(a, bb, acc);
    }
    __syncthreads();
  }
  int n = n0 + (lane & 15);
  int mb = (lane >> 4) << 3;
  bool real = (n < 64);
  int c = real ? n : n - 64;
  float g  = real ? gr[c] : gi[c];
  float be = real ? br[c] : bi[c];
  float mu = real ? mr[c] : mi[c];
  float va = real ? vr[c] : vi[c];
  float sc = g * rsqrtf(va + EPS);
  h16_t* dst = real ? xr : xi;
#pragma unroll
  for (int i = 0; i < 8; ++i)
    dst[(size_t)(mt * 16 + mb + i) * 128 + c] = (h16_t)(sc * (acc[i] - mu) + be);
}

// ---------------- GRU input projections: xp = x @ k + b[0] ------------------
// grid (384, 4), block 256; s selects (x, weight, bias) combo.
__global__ void proj_kernel(const h16_t* __restrict__ xr, const h16_t* __restrict__ xi,
                            const h16_t* __restrict__ gk, const float* __restrict__ brr,
                            const float* __restrict__ bii, int Dl,
                            h16_t* __restrict__ xp) {
  int mt = blockIdx.x, s = blockIdx.y;
  int tid = threadIdx.x, wv = tid >> 5, lane = tid & 31;
  const h16_t* x = (s == 1 || s == 3) ? xi : xr;
  const h16_t* wgt = gk + ((s >= 2) ? (size_t)128 * TU : 0);
  const float* bias = (s >= 2) ? bii : brr;             // row 0 = b[0]
  size_t row0 = (size_t)mt * 16;
  v8f acc[3] = {};
  int kts = Dl / 32;
  for (int kt = 0; kt < kts; ++kt) {
    v16h a = load_a(x + row0 * 128 + kt * 32, 128);
#pragma unroll
    for (int j = 0; j < 3; ++j) {
      int nt = wv + j * 8;
      v16h bb = load_bfrag(wgt + (size_t)(kt * 24 + nt) * 512);
      acc[j] = wmma32(a, bb, acc[j]);
    }
  }
  int mb = (lane >> 4) << 3;
  h16_t* out = xp + (size_t)s * BT * TU;
#pragma unroll
  for (int j = 0; j < 3; ++j) {
    int n = (wv + j * 8) * 16 + (lane & 15);
    float b0 = bias[n];
#pragma unroll
    for (int i = 0; i < 8; ++i)
      out[(row0 + mb + i) * TU + n] = (h16_t)(acc[j][i] + b0);
  }
}

// ---------------- recurrent scan (the latency-critical part) ----------------
// grid 4 (one persistent WG per sequence), block 256 (8 waves).
// rk WMMA B-fragments live in VGPRs; h lives in LDS; 2 barriers / step.
__global__ void scan_kernel(const h16_t* __restrict__ xp, const h16_t* __restrict__ grk,
                            const float* __restrict__ brr, const float* __restrict__ bii,
                            h16_t* __restrict__ ys) {
  int s = blockIdx.x;
  int tid = threadIdx.x, wv = tid >> 5;
  const h16_t* rk = grk + ((s >= 2) ? (size_t)128 * TU : 0);
  const float* b1 = ((s >= 2) ? bii : brr) + TU;        // b[1]
  const h16_t* xps = xp + (size_t)s * BT * TU;
  h16_t* yss = ys + (size_t)s * BT * 128;

  __shared__ __align__(32) h16_t hh[16 * 128];   // hidden state, f16 (WMMA A)
  __shared__ __align__(32) float hf[16 * 128];   // hidden state, f32 master
  __shared__ __align__(32) float rec[16 * TU];   // h @ rk

  for (int i = tid; i < 16 * 128; i += 256) { hf[i] = 0.f; hh[i] = (h16_t)0.f; }

  // resident recurrent-weight fragments: 3 n-tiles x 4 k-tiles per wave
  v16h Bf[3][4];
#pragma unroll
  for (int j = 0; j < 3; ++j) {
    int nt = wv + j * 8;
#pragma unroll
    for (int kt = 0; kt < 4; ++kt)
      Bf[j][kt] = load_bfrag(rk + (size_t)(kt * 24 + nt) * 512);
  }
  __syncthreads();

  int r = tid >> 4;                 // batch row handled in gate phase
  int u0 = (tid & 15) * 8;

  for (int t = 0; t < T_; ++t) {
    // rec = h @ rk   (16x384x128 per step, 12 WMMA per wave)
    v16h a[4];
#pragma unroll
    for (int kt = 0; kt < 4; ++kt) a[kt] = load_a(hh + kt * 32, 128);
#pragma unroll
    for (int j = 0; j < 3; ++j) {
      v8f acc = {};
#pragma unroll
      for (int kt = 0; kt < 4; ++kt) acc = wmma32(a[kt], Bf[j][kt], acc);
      store_c(rec + (wv + j * 8) * 16, TU, acc);
    }
    __syncthreads();

    // gates (keras reset_after): z,r sigmoid; hh = tanh(xh + r*(rec_h+b1_h))
    const h16_t* xrow = xps + (size_t)(r * T_ + t) * TU;
    if (t + 8 < T_) __builtin_prefetch(xps + (size_t)(r * T_ + t + 8) * TU, 0, 1);
    v8h xzv = *(const v8h*)(xrow + u0);
    v8h xrv = *(const v8h*)(xrow + U_ + u0);
    v8h xhv = *(const v8h*)(xrow + 2 * U_ + u0);
    v8h hnv;
#pragma unroll
    for (int q = 0; q < 8; ++q) {
      int u = u0 + q;
      float rz = rec[r * TU + u] + b1[u];
      float rr_ = rec[r * TU + U_ + u] + b1[U_ + u];
      float rh = rec[r * TU + 2 * U_ + u] + b1[2 * U_ + u];
      float z = 1.f / (1.f + __expf(-((float)xzv[q] + rz)));
      float rg = 1.f / (1.f + __expf(-((float)xrv[q] + rr_)));
      float ht = tanhf((float)xhv[q] + rg * rh);
      float hn = z * hf[r * 128 + u] + (1.f - z) * ht;
      hf[r * 128 + u] = hn;
      hh[r * 128 + u] = (h16_t)hn;
      hnv[q] = (h16_t)hn;
    }
    *(v8h*)(yss + (size_t)(r * T_ + t) * 128 + u0) = hnv;
    __syncthreads();
  }
}

// next_xr = rr - ii ; next_xi = ri + ir   (vectorized, 8 halves / thread)
__global__ void combine_kernel(const h16_t* __restrict__ ys,
                               h16_t* __restrict__ xr, h16_t* __restrict__ xi) {
  int v = blockIdx.x * 256 + threadIdx.x;
  if (v >= (BT * 128) / 8) return;
  size_t off = (size_t)v * 8;
  v8h s0 = *(const v8h*)(ys + off);
  v8h s1 = *(const v8h*)(ys + (size_t)BT * 128 + off);
  v8h s2 = *(const v8h*)(ys + (size_t)2 * BT * 128 + off);
  v8h s3 = *(const v8h*)(ys + (size_t)3 * BT * 128 + off);
  v8h orr, oii;
#pragma unroll
  for (int q = 0; q < 8; ++q) {
    orr[q] = (h16_t)((float)s0[q] - (float)s3[q]);
    oii[q] = (h16_t)((float)s1[q] + (float)s2[q]);
  }
  *(v8h*)(xr + off) = orr;
  *(v8h*)(xi + off) = oii;
}

// ---------------- FF block: y = x + relu(x@W1+b1)@W2+b2 --------------------
// grid (384, 2), block 256; t1 tile staged in LDS between the two GEMMs.
__global__ void ff_kernel(const h16_t* __restrict__ xrr, const h16_t* __restrict__ xii,
                          const h16_t* __restrict__ w1, const h16_t* __restrict__ w2,
                          const float* __restrict__ b1, const float* __restrict__ b2,
                          float* __restrict__ out) {
  __shared__ __align__(32) h16_t t1[16 * FF_];
  int mt = blockIdx.x, p = blockIdx.y;
  const h16_t* x = p ? xii : xrr;
  float* o = out + (size_t)p * BT * 128;
  int tid = threadIdx.x, wv = tid >> 5, lane = tid & 31;
  size_t row0 = (size_t)mt * 16;

  v8f acc[8] = {};
#pragma unroll
  for (int kt = 0; kt < 4; ++kt) {
    v16h a = load_a(x + row0 * 128 + kt * 32, 128);
#pragma unroll
    for (int j = 0; j < 8; ++j) {
      int nt = wv * 8 + j;
      v16h bb = load_bfrag(w1 + (size_t)(kt * 64 + nt) * 512);
      acc[j] = wmma32(a, bb, acc[j]);
    }
  }
  int mb = (lane >> 4) << 3;
#pragma unroll
  for (int j = 0; j < 8; ++j) {
    int n = (wv * 8 + j) * 16 + (lane & 15);
    float bb1 = b1[n];
#pragma unroll
    for (int i = 0; i < 8; ++i) {
      float v = acc[j][i] + bb1;
      t1[(size_t)(mb + i) * FF_ + n] = (h16_t)(v > 0.f ? v : 0.f);
    }
  }
  __syncthreads();

  v8f a2 = {};
  for (int kt = 0; kt < 32; ++kt) {
    v16h a = load_a(t1 + kt * 32, FF_);
    v16h bb = load_bfrag(w2 + (size_t)(kt * 8 + wv) * 512);
    a2 = wmma32(a, bb, a2);
  }
  int n = wv * 16 + (lane & 15);
  float bb2 = b2[n];
#pragma unroll
  for (int i = 0; i < 8; ++i) {
    size_t rr = row0 + mb + i;
    o[rr * 128 + n] = a2[i] + bb2 + (float)x[rr * 128 + n];
  }
}

// ---------------- time mean ------------------------------------------------
__global__ void mean_kernel(const float* __restrict__ ffout,
                            float* __restrict__ mmean, h16_t* __restrict__ m16) {
  int p = blockIdx.x >> 4, b = blockIdx.x & 15, u = threadIdx.x;
  const float* src = ffout + (size_t)p * BT * 128 + (size_t)b * T_ * 128 + u;
  float s = 0.f;
  for (int t = 0; t < T_; ++t) s += src[(size_t)t * 128];
  s *= (1.f / T_);
  size_t idx = (size_t)p * 16 * 128 + b * 128 + u;
  mmean[idx] = s;
  m16[idx] = (h16_t)s;
}

// ---------------- dense heads: Q (to qtmp), R (to d_out) -------------------
// grid 32, block 256 -> 256 N-tiles of 16 over NN=4096.
__global__ void dense_kernel(const h16_t* __restrict__ m16, const h16_t* __restrict__ qw,
                             const float* __restrict__ qrb, const float* __restrict__ qib,
                             const float* __restrict__ rrb, const float* __restrict__ rib,
                             float* __restrict__ qtmp, float* __restrict__ out) {
  int tid = threadIdx.x, wv = tid >> 5, lane = tid & 31;
  int ntg = blockIdx.x * 8 + wv;                 // global N-tile (0..255)
  int n0 = ntg * 16;
  const h16_t* mr = m16;
  const h16_t* mi = m16 + 16 * 128;
  const h16_t* wqr = qw;
  const h16_t* wqi = qw + SZ_QW1;
  const h16_t* wrr = qw + 2 * SZ_QW1;
  const h16_t* wri = qw + 3 * SZ_QW1;
  v8f q1 = {}, q2 = {}, q3 = {}, q4 = {}, r1 = {}, r2 = {}, r3 = {}, r4 = {};
#pragma unroll
  for (int kt = 0; kt < 4; ++kt) {
    v16h ar = load_a(mr + kt * 32, 128);
    v16h ai = load_a(mi + kt * 32, 128);
    size_t tile = (size_t)(kt * 256 + ntg) * 512;
    v16h b;
    b = load_bfrag(wqr + tile);
    q1 = wmma32(ar, b, q1); q3 = wmma32(ai, b, q3);
    b = load_bfrag(wqi + tile);
    q2 = wmma32(ai, b, q2); q4 = wmma32(ar, b, q4);
    b = load_bfrag(wrr + tile);
    r1 = wmma32(ar, b, r1); r3 = wmma32(ai, b, r3);
    b = load_bfrag(wri + tile);
    r2 = wmma32(ai, b, r2); r4 = wmma32(ar, b, r4);
  }
  int n = n0 + (lane & 15), mb = (lane >> 4) << 3;
  float bqr = qrb[n], bqi = qib[n], brr = rrb[n], bri = rib[n];
#pragma unroll
  for (int i = 0; i < 8; ++i) {
    int bidx = mb + i;                               // batch index
    qtmp[(size_t)bidx * NN + n]              = q1[i] - q2[i] + bqr - bqi;  // Qr
    qtmp[(size_t)16 * NN + bidx * NN + n]    = q3[i] + q4[i] + bqr + bqi;  // Qi
    out[(size_t)2 * 16 * NN + bidx * NN + n] = r1[i] - r2[i] + brr - bri;  // Rr
    out[(size_t)3 * 16 * NN + bidx * NN + n] = r3[i] + r4[i] + brr + bri;  // Ri
  }
}

__global__ void norm_kernel(const float* __restrict__ qtmp, float* __restrict__ scale) {
  __shared__ float red[256];
  int b = blockIdx.x;
  float s = 0.f;
  for (int n = threadIdx.x; n < NN; n += 256) {
    float a = qtmp[(size_t)b * NN + n];
    float c = qtmp[(size_t)16 * NN + b * NN + n];
    s += a * a + c * c;
  }
  red[threadIdx.x] = s;
  __syncthreads();
  for (int st = 128; st; st >>= 1) {
    if (threadIdx.x < st) red[threadIdx.x] += red[threadIdx.x + st];
    __syncthreads();
  }
  if (threadIdx.x == 0) scale[b] = sqrtf(64.f) / sqrtf(red[0]);
}

__global__ void scaleq_kernel(const float* __restrict__ qtmp,
                              const float* __restrict__ scale, float* __restrict__ out) {
  int i = blockIdx.x * 256 + threadIdx.x;
  if (i >= 2 * 16 * NN) return;
  int b = (i % (16 * NN)) / NN;
  out[i] = qtmp[i] * scale[b];
}

// ---------------------------------------------------------------------------
extern "C" void kernel_launch(void* const* d_in, const int* in_sizes, int n_in,
                              void* d_out, int out_size, void* d_ws, size_t ws_size,
                              hipStream_t stream) {
  (void)in_sizes; (void)n_in; (void)out_size; (void)ws_size;
  const float* h_real = (const float*)d_in[0];
  const float* h_imag = (const float*)d_in[1];
  const float* conv_wr = (const float*)d_in[2];
  const float* conv_wi = (const float*)d_in[3];
  const float* bn_gr = (const float*)d_in[4];
  const float* bn_gi = (const float*)d_in[5];
  const float* bn_vr = (const float*)d_in[6];
  const float* bn_vi = (const float*)d_in[7];
  const float* bn_br = (const float*)d_in[8];
  const float* bn_bi = (const float*)d_in[9];
  const float* bn_mr = (const float*)d_in[10];
  const float* bn_mi = (const float*)d_in[11];
  const float* g0_kr = (const float*)d_in[12];
  const float* g0_rkr = (const float*)d_in[13];
  const float* g0_br = (const float*)d_in[14];
  const float* g0_ki = (const float*)d_in[15];
  const float* g0_rki = (const float*)d_in[16];
  const float* g0_bi = (const float*)d_in[17];
  const float* gs_kr = (const float*)d_in[18];
  const float* gs_rkr = (const float*)d_in[19];
  const float* gs_br = (const float*)d_in[20];
  const float* gs_ki = (const float*)d_in[21];
  const float* gs_rki = (const float*)d_in[22];
  const float* gs_bi = (const float*)d_in[23];
  const float* ff1_w = (const float*)d_in[24];
  const float* ff1_b = (const float*)d_in[25];
  const float* ff2_w = (const float*)d_in[26];
  const float* ff2_b = (const float*)d_in[27];
  const float* qr_w = (const float*)d_in[28];
  const float* qr_b = (const float*)d_in[29];
  const float* qi_w = (const float*)d_in[30];
  const float* qi_b = (const float*)d_in[31];
  const float* rr_w = (const float*)d_in[32];
  const float* rr_b = (const float*)d_in[33];
  const float* ri_w = (const float*)d_in[34];
  const float* ri_b = (const float*)d_in[35];

  h16_t* wsH = (h16_t*)d_ws;
  h16_t* Wc   = wsH + OFF_WC16;
  h16_t* Hcat = wsH + OFF_HCAT;
  h16_t* GK   = wsH + OFF_GK;
  h16_t* GRK  = wsH + OFF_GRK;
  h16_t* FF1h = wsH + OFF_FF1;
  h16_t* FF2h = wsH + OFF_FF2;
  h16_t* QW   = wsH + OFF_QW;
  h16_t* XR   = wsH + OFF_XR;
  h16_t* XI   = wsH + OFF_XI;
  h16_t* XP   = wsH + OFF_XP;
  h16_t* YS   = wsH + OFF_YS;
  h16_t* M16  = wsH + OFF_M16;
  float* wsF = (float*)((char*)d_ws + ((HALF_TOTAL * 2 + 255) / 256 * 256));
  float* FFOUT = wsF;
  float* MMEAN = FFOUT + 2ull * BT * 128;
  float* QTMP  = MMEAN + 2ull * 16 * 128;
  float* SCL   = QTMP + 2ull * 16 * NN;
  float* OUT = (float*)d_out;

  auto g = [](int n) { return (n + 255) / 256; };

  // ---- weight / input packing to fragment-major f16 ----
  pack_conv_frag<<<g(KW * 128 * 128), 256, 0, stream>>>(conv_wr, conv_wi, Wc);
  pack_hcat<<<g(BT * 128), 256, 0, stream>>>(h_real, h_imag, Hcat);
  pack_frag<<<g(64 * TU), 256, 0, stream>>>(g0_kr, GK, 64, TU, 1, 0, 0);
  pack_frag<<<g(64 * TU), 256, 0, stream>>>(g0_ki, GK + 128 * TU, 64, TU, 1, 0, 0);
  pack_frag<<<g(128 * TU), 256, 0, stream>>>(g0_rkr, GRK, 128, TU, 1, 0, 0);
  pack_frag<<<g(128 * TU), 256, 0, stream>>>(g0_rki, GRK + 128 * TU, 128, TU, 1, 0, 0);
  int nls = LS * 128 * TU;
  pack_frag<<<g(nls), 256, 0, stream>>>(gs_kr,  GK + SZ_GK_L,            128, TU, LS, (size_t)128 * TU, SZ_GK_L);
  pack_frag<<<g(nls), 256, 0, stream>>>(gs_ki,  GK + SZ_GK_L + 128 * TU, 128, TU, LS, (size_t)128 * TU, SZ_GK_L);
  pack_frag<<<g(nls), 256, 0, stream>>>(gs_rkr, GRK + SZ_GK_L,           128, TU, LS, (size_t)128 * TU, SZ_GK_L);
  pack_frag<<<g(nls), 256, 0, stream>>>(gs_rki, GRK + SZ_GK_L + 128 * TU, 128, TU, LS, (size_t)128 * TU, SZ_GK_L);
  pack_frag<<<g(128 * FF_), 256, 0, stream>>>(ff1_w, FF1h, 128, FF_, 1, 0, 0);
  pack_frag<<<g(FF_ * 128), 256, 0, stream>>>(ff2_w, FF2h, FF_, 128, 1, 0, 0);
  pack_frag<<<g(128 * NN), 256, 0, stream>>>(qr_w, QW, 128, NN, 1, 0, 0);
  pack_frag<<<g(128 * NN), 256, 0, stream>>>(qi_w, QW + SZ_QW1, 128, NN, 1, 0, 0);
  pack_frag<<<g(128 * NN), 256, 0, stream>>>(rr_w, QW + 2 * SZ_QW1, 128, NN, 1, 0, 0);
  pack_frag<<<g(128 * NN), 256, 0, stream>>>(ri_w, QW + 3 * SZ_QW1, 128, NN, 1, 0, 0);

  // ---- conv + BN ----
  conv_bn_kernel<<<384, 256, 0, stream>>>(Hcat, Wc, bn_gr, bn_br, bn_mr, bn_vr,
                                          bn_gi, bn_bi, bn_mi, bn_vi, XR, XI);

  // ---- 16 complex GRU layers (sequential) ----
  for (int l = 0; l < NLAYER; ++l) {
    const h16_t* gk = GK + (size_t)l * SZ_GK_L;
    const h16_t* grk = GRK + (size_t)l * SZ_GK_L;
    const float* br = l ? gs_br + (size_t)(l - 1) * 2 * TU : g0_br;
    const float* bi = l ? gs_bi + (size_t)(l - 1) * 2 * TU : g0_bi;
    int Dl = l ? 128 : 64;
    proj_kernel<<<dim3(384, 4), 256, 0, stream>>>(XR, XI, gk, br, bi, Dl, XP);
    scan_kernel<<<4, 256, 0, stream>>>(XP, grk, br, bi, YS);
    combine_kernel<<<g(BT * 128 / 8), 256, 0, stream>>>(YS, XR, XI);
  }

  // ---- FF + residual, mean, heads, norm ----
  ff_kernel<<<dim3(384, 2), 256, 0, stream>>>(XR, XI, FF1h, FF2h, ff1_b, ff2_b, FFOUT);
  mean_kernel<<<32, 128, 0, stream>>>(FFOUT, MMEAN, M16);
  dense_kernel<<<32, 256, 0, stream>>>(M16, QW, qr_b, qi_b, rr_b, ri_b, QTMP, OUT);
  norm_kernel<<<16, 256, 0, stream>>>(QTMP, SCL);
  scaleq_kernel<<<g(2 * 16 * NN), 256, 0, stream>>>(QTMP, SCL, OUT);
}
